// TriangleAttention_64707977281608
// MI455X (gfx1250) — matrix operand order repaired and checked by reading, not compiled, not run on hardware
//
#include <hip/hip_runtime.h>
#include <hip/hip_bf16.h>
#include <math.h>

typedef _Float16 v16h __attribute__((ext_vector_type(16)));
typedef _Float16 half8 __attribute__((ext_vector_type(8)));
typedef _Float16 half4 __attribute__((ext_vector_type(4)));
typedef float    v8f  __attribute__((ext_vector_type(8)));
typedef float    f4   __attribute__((ext_vector_type(4)));

#define NPIX   65536      // 256*256 pixels
#define CIN    128
#define CHID   32
#define NHEAD  4
#define NSEQ   256
#define QSCALE 0.17677669529663689f   // 1/sqrt(32)

// ---------------------------------------------------------------------------
// Fragment helpers (V_WMMA_F32_16X16X32_F16 layouts, wave32)
// A (16x32 f16): lane m=lane&15; halves t<8 -> K = 8*g + t ; t>=8 -> K = 16+8*g+(t-8)
// B (32x16 f16): lane n=lane&15; halves t<8 -> K = 8*g + t ; t>=8 -> K = 16+8*g+(t-8)
// C (16x16 f32): col = lane&15 ; VGPR r -> row = r + 8*g
// ---------------------------------------------------------------------------
__device__ __forceinline__ v16h make_frag(half8 lo, half8 hi) {
  v16h r;
#pragma unroll
  for (int t = 0; t < 8; ++t) { r[t] = lo[t]; r[t + 8] = hi[t]; }
  return r;
}

// base_row = &src[(m_row)*ld + k0]   (row-major, this lane's row)
__device__ __forceinline__ v16h load_a16(const _Float16* base_row, int g) {
  half8 lo = *(const half8*)(base_row + 8 * g);
  half8 hi = *(const half8*)(base_row + 16 + 8 * g);
  return make_frag(lo, hi);
}

// base_col = &src[(n_col)*ld + k0]   ("n-major" storage: contiguous along K)
__device__ __forceinline__ v16h load_b16(const _Float16* base_col, int g) {
  half8 lo = *(const half8*)(base_col + 8 * g);
  half8 hi = *(const half8*)(base_col + 16 + 8 * g);
  return make_frag(lo, hi);
}

__device__ __forceinline__ v8f wmma_f16(v16h a, v16h b, v8f c) {
  return __builtin_amdgcn_wmma_f32_16x16x32_f16(false, a, false, b, (short)0, c,
                                                false, false);
}

__device__ __forceinline__ float wred_sum(float v) {
#pragma unroll
  for (int m = 1; m < 32; m <<= 1) v += __shfl_xor(v, m, 32);
  return v;
}

// ---------------------------------------------------------------------------
// 1) LayerNorm (f32 -> f16) + triangle bias tri[h][pixel] = xn(pixel) . w_b[h]
//    one wave per pixel (32 lanes x 4 channels)
// ---------------------------------------------------------------------------
__global__ __launch_bounds__(256) void ln_kernel(const float* __restrict__ x,
                                                 const float* __restrict__ ln_w,
                                                 const float* __restrict__ ln_b,
                                                 const float* __restrict__ w_b,
                                                 _Float16* __restrict__ xn16,
                                                 float* __restrict__ tri) {
  const int lane = threadIdx.x & 31;
  const int w    = threadIdx.x >> 5;
  const int pix  = blockIdx.x * 8 + w;

  f4 xv = *(const f4*)(x + (size_t)pix * CIN + lane * 4);
  float s  = xv[0] + xv[1] + xv[2] + xv[3];
  float s2 = xv[0]*xv[0] + xv[1]*xv[1] + xv[2]*xv[2] + xv[3]*xv[3];
  s  = wred_sum(s);
  s2 = wred_sum(s2);
  const float mean = s * (1.0f / CIN);
  const float var  = s2 * (1.0f / CIN) - mean * mean;
  const float rstd = rsqrtf(var + 1e-5f);

  float xn[4];
  half4 hv;
#pragma unroll
  for (int c = 0; c < 4; ++c) {
    const int ch = lane * 4 + c;
    xn[c] = (xv[c] - mean) * rstd * ln_w[ch] + ln_b[ch];
    hv[c] = (_Float16)xn[c];
  }
  *(half4*)(xn16 + (size_t)pix * CIN + lane * 4) = hv;

#pragma unroll
  for (int hh = 0; hh < NHEAD; ++hh) {
    const float* wb = w_b + hh * CIN + lane * 4;
    float d = xn[0]*wb[0] + xn[1]*wb[1] + xn[2]*wb[2] + xn[3]*wb[3];
    d = wred_sum(d);
    if (lane == 0) tri[(size_t)hh * NPIX + pix] = d;
  }
}

// ---------------------------------------------------------------------------
// 2) Weight conversion f32 -> f16 (layouts already [n][k], B-fragment friendly)
//    Wqkvg: [512][128]  (q | k | v | g stacked), Wo: [128][128]
// ---------------------------------------------------------------------------
__global__ __launch_bounds__(256) void cvt_kernel(const float* __restrict__ wq,
                                                  const float* __restrict__ wk,
                                                  const float* __restrict__ wv,
                                                  const float* __restrict__ wg,
                                                  const float* __restrict__ wo,
                                                  _Float16* __restrict__ Wqkvg,
                                                  _Float16* __restrict__ Wo) {
  const int idx = blockIdx.x * 256 + threadIdx.x;
  if (idx < 512 * CIN) {
    const int n = idx >> 7, k = idx & 127;
    const int sec = n >> 7, nn = n & 127;
    const float* src = (sec == 0) ? wq : (sec == 1) ? wk : (sec == 2) ? wv : wg;
    Wqkvg[idx] = (_Float16)src[nn * CIN + k];
  } else if (idx < 512 * CIN + CIN * CIN) {
    const int j = idx - 512 * CIN;
    Wo[j] = (_Float16)wo[j];
  }
}

// ---------------------------------------------------------------------------
// 3) Fused QKVG projection GEMM: [65536 x 128] x [128 x 512]
//    one 16x16 output tile per wave; K = 128 -> 4 WMMAs
//    epilogue: q scaled, k plain, v transposed [ih][d][j], g sigmoid
// ---------------------------------------------------------------------------
__global__ __launch_bounds__(256) void proj_kernel(const _Float16* __restrict__ xn,
                                                   const _Float16* __restrict__ W,
                                                   const float* __restrict__ b_g,
                                                   _Float16* __restrict__ qb,
                                                   _Float16* __restrict__ kb,
                                                   _Float16* __restrict__ vTb,
                                                   _Float16* __restrict__ gb) {
  const int lane = threadIdx.x & 31;
  const int wv_  = threadIdx.x >> 5;
  const int tile = blockIdx.x * 8 + wv_;
  const int mt = tile >> 5, nt = tile & 31;
  const int m0 = mt * 16, n0 = nt * 16;
  const int g = lane >> 4, nloc = lane & 15;

  v8f acc = {};
#pragma unroll
  for (int kk = 0; kk < CIN; kk += 32) {
    v16h a = load_a16(xn + (size_t)(m0 + nloc) * CIN + kk, g);
    v16h b = load_b16(W  + (size_t)(n0 + nloc) * CIN + kk, g);
    acc = wmma_f16(a, b, acc);
  }

  const int n = n0 + nloc;
  const int sec = n >> 7;            // 0=q 1=k 2=v 3=g  (uniform per wave)
  const int e = n & 127, h = e >> 5, d = e & 31;
  const float bg = (sec == 3) ? b_g[e] : 0.0f;

#pragma unroll
  for (int r = 0; r < 8; ++r) {
    const int m = m0 + r + 8 * g;     // pixel index
    const int i = m >> 8, j = m & 255;
    const float val = acc[r];
    const size_t ihjd = ((size_t)(i * NHEAD + h) * NSEQ + j) * CHID + d;
    if (sec == 0) {
      qb[ihjd] = (_Float16)(val * QSCALE);
    } else if (sec == 1) {
      kb[ihjd] = (_Float16)val;
    } else if (sec == 2) {
      vTb[((size_t)(i * NHEAD + h) * CHID + d) * NSEQ + j] = (_Float16)val;
    } else {
      const float sg = val + bg;
      gb[(size_t)m * CIN + e] = (_Float16)(1.0f / (1.0f + __expf(-sg)));
    }
  }
}

// ---------------------------------------------------------------------------
// 4) Attention: one workgroup per (i,h); 8 waves x 2 q-tiles of 16 queries.
//    scores (16 WMMAs) in registers -> bias -> softmax (shuffle over 16-lane
//    halves, matching C layout) -> probs to per-wave LDS slab -> P.V
//    (2 d-tiles x 8 accumulating WMMAs) -> fused sigmoid gating -> og f16
// ---------------------------------------------------------------------------
__global__ __launch_bounds__(256) void attn_kernel(const _Float16* __restrict__ qb,
                                                   const _Float16* __restrict__ kb,
                                                   const _Float16* __restrict__ vTb,
                                                   const _Float16* __restrict__ gb,
                                                   const float* __restrict__ tri,
                                                   const float* __restrict__ mask,
                                                   _Float16* __restrict__ og) {
  __shared__ _Float16 probs[8][16][NSEQ];   // 64 KB, per-wave slabs

  const int lane = threadIdx.x & 31;
  const int w    = threadIdx.x >> 5;
  const int g    = lane >> 4, nloc = lane & 15;
  const int i = blockIdx.x >> 2, h = blockIdx.x & 3;
  const int ih = i * NHEAD + h;

  const _Float16* qp = qb  + (size_t)ih * NSEQ * CHID;
  const _Float16* kp = kb  + (size_t)ih * NSEQ * CHID;
  const _Float16* vp = vTb + (size_t)ih * CHID * NSEQ;
  const float*    tp = tri + (size_t)h * NPIX;

  for (int it = 0; it < 2; ++it) {
    const int q0 = (w + it * 8) * 16;

    // ---- scores: QK^T (K = 32, one WMMA per key tile) + biases -------------
    v16h aq = load_a16(qp + (size_t)(q0 + nloc) * CHID, g);
    v8f sc[16];
#pragma unroll
    for (int kt = 0; kt < 16; ++kt) {
      v16h bk = load_b16(kp + (size_t)(kt * 16 + nloc) * CHID, g);
      v8f c = {};
      c = wmma_f16(aq, bk, c);
      const int key = kt * 16 + nloc;
      const float mb = 1e9f * (mask[i * NSEQ + key] - 1.0f);
      const float* tcol = tp + key;
#pragma unroll
      for (int r = 0; r < 8; ++r) {
        const int qrow = q0 + r + 8 * g;
        c[r] += mb + tcol[(size_t)qrow * NSEQ];
      }
      sc[kt] = c;
    }

    // ---- softmax over 256 keys (rows split across lane halves) -------------
    v8f mx = sc[0];
#pragma unroll
    for (int kt = 1; kt < 16; ++kt)
#pragma unroll
      for (int r = 0; r < 8; ++r) mx[r] = fmaxf(mx[r], sc[kt][r]);
#pragma unroll
    for (int r = 0; r < 8; ++r) {
      float v = mx[r];
      v = fmaxf(v, __shfl_xor(v, 1, 32));
      v = fmaxf(v, __shfl_xor(v, 2, 32));
      v = fmaxf(v, __shfl_xor(v, 4, 32));
      v = fmaxf(v, __shfl_xor(v, 8, 32));
      mx[r] = v;
    }
    v8f sm = {};
#pragma unroll
    for (int kt = 0; kt < 16; ++kt)
#pragma unroll
      for (int r = 0; r < 8; ++r) {
        sc[kt][r] = __expf(sc[kt][r] - mx[r]);
        sm[r] += sc[kt][r];
      }
#pragma unroll
    for (int r = 0; r < 8; ++r) {
      float v = sm[r];
      v += __shfl_xor(v, 1, 32);
      v += __shfl_xor(v, 2, 32);
      v += __shfl_xor(v, 4, 32);
      v += __shfl_xor(v, 8, 32);
      sm[r] = 1.0f / v;
    }

    // ---- probs -> per-wave LDS slab (row-major f16) ------------------------
#pragma unroll
    for (int kt = 0; kt < 16; ++kt)
#pragma unroll
      for (int r = 0; r < 8; ++r)
        probs[w][r + 8 * g][kt * 16 + nloc] = (_Float16)(sc[kt][r] * sm[r]);

    // ---- P.V : K = 256 keys -> 8 accumulating WMMAs per 16-d tile ---------
#pragma unroll
    for (int dt = 0; dt < 2; ++dt) {
      v8f acc = {};
#pragma unroll
      for (int kc = 0; kc < 8; ++kc) {
        v16h ap = load_a16(&probs[w][nloc][kc * 32], g);
        v16h bv = load_b16(vp + (size_t)(dt * 16 + nloc) * NSEQ + kc * 32, g);
        acc = wmma_f16(ap, bv, acc);
      }
      const int dcol = dt * 16 + nloc;
      const int e = h * CHID + dcol;
#pragma unroll
      for (int r = 0; r < 8; ++r) {
        const int pix = i * NSEQ + q0 + r + 8 * g;
        const float gate = (float)gb[(size_t)pix * CIN + e];
        og[(size_t)pix * CIN + e] = (_Float16)(acc[r] * gate);
      }
    }
  }
}

// ---------------------------------------------------------------------------
// 5) Output projection: [65536 x 128] x [128 x 128] + b_o -> f32 out
// ---------------------------------------------------------------------------
__global__ __launch_bounds__(256) void out_kernel(const _Float16* __restrict__ og,
                                                  const _Float16* __restrict__ Wo,
                                                  const float* __restrict__ b_o,
                                                  float* __restrict__ out) {
  const int lane = threadIdx.x & 31;
  const int wv_  = threadIdx.x >> 5;
  const int tile = blockIdx.x * 8 + wv_;
  const int mt = tile >> 3, nt = tile & 7;
  const int m0 = mt * 16, n0 = nt * 16;
  const int g = lane >> 4, nloc = lane & 15;

  v8f acc = {};
#pragma unroll
  for (int kk = 0; kk < CIN; kk += 32) {
    v16h a = load_a16(og + (size_t)(m0 + nloc) * CIN + kk, g);
    v16h b = load_b16(Wo + (size_t)(n0 + nloc) * CIN + kk, g);
    acc = wmma_f16(a, b, acc);
  }
  const float bias = b_o[n0 + nloc];
#pragma unroll
  for (int r = 0; r < 8; ++r) {
    const int m = m0 + r + 8 * g;
    out[(size_t)m * CIN + n0 + nloc] = acc[r] + bias;
  }
}

// ---------------------------------------------------------------------------
extern "C" void kernel_launch(void* const* d_in, const int* in_sizes, int n_in,
                              void* d_out, int out_size, void* d_ws, size_t ws_size,
                              hipStream_t stream) {
  const float* x    = (const float*)d_in[0];
  const float* mask = (const float*)d_in[1];
  const float* ln_w = (const float*)d_in[2];
  const float* ln_b = (const float*)d_in[3];
  const float* w_b  = (const float*)d_in[4];
  const float* w_q  = (const float*)d_in[5];
  const float* w_k  = (const float*)d_in[6];
  const float* w_v  = (const float*)d_in[7];
  const float* w_g  = (const float*)d_in[8];
  const float* b_g  = (const float*)d_in[9];
  const float* w_o  = (const float*)d_in[10];
  const float* b_o  = (const float*)d_in[11];
  float* out = (float*)d_out;

  char* ws = (char*)d_ws;
  const size_t SZ16 = (size_t)NPIX * CIN * sizeof(_Float16);   // 16 MB
  _Float16* xn16  = (_Float16*)(ws);                 // reused as og after proj
  _Float16* q16   = (_Float16*)(ws + SZ16);
  _Float16* k16   = (_Float16*)(ws + 2 * SZ16);
  _Float16* vT16  = (_Float16*)(ws + 3 * SZ16);
  _Float16* g16   = (_Float16*)(ws + 4 * SZ16);
  _Float16* Wqkvg = (_Float16*)(ws + 5 * SZ16);
  _Float16* Wo16  = (_Float16*)(ws + 5 * SZ16 + 512 * CIN * sizeof(_Float16));
  float*    tri   = (float*)   (ws + 5 * SZ16 + (512 + 128) * CIN * sizeof(_Float16));
  _Float16* og16  = xn16;   // alias: xn dead after projection GEMM

  ln_kernel<<<NPIX / 8, 256, 0, stream>>>(x, ln_w, ln_b, w_b, xn16, tri);
  cvt_kernel<<<(512 * CIN + CIN * CIN + 255) / 256, 256, 0, stream>>>(
      w_q, w_k, w_v, w_g, w_o, Wqkvg, Wo16);
  proj_kernel<<<(NPIX / 16) * 32 / 8, 256, 0, stream>>>(xn16, Wqkvg, b_g, q16,
                                                        k16, vT16, g16);
  attn_kernel<<<NSEQ * NHEAD, 256, 0, stream>>>(q16, k16, vT16, g16, tri, mask,
                                                og16);
  out_kernel<<<(NPIX / 16) * 8 / 8, 256, 0, stream>>>(og16, Wo16, b_o, out);
}
